// ResGCN_36893769072894
// MI455X (gfx1250) — matrix operand order repaired and checked by reading, not compiled, not run on hardware
//
#include <hip/hip_runtime.h>

typedef _Float16 v16h __attribute__((ext_vector_type(16)));
typedef float    v8f  __attribute__((ext_vector_type(8)));

constexpr int   NPTS      = 16384;
constexpr int   BATCH     = 2;
constexpr int   KNBR      = 16;
constexpr float EPS       = 1e-5f;
constexpr int   GEMM_GRID = 4096;   // single-wave blocks; 4 points per wave

// ---------------------------------------------------------------------------
// Extract x (N x 4) from points (N x 5, col 0 = batch idx)
// ---------------------------------------------------------------------------
__global__ void extract_x(const float* __restrict__ pts, float* __restrict__ x4) {
    int i = blockIdx.x * 256 + threadIdx.x;
    if (i < NPTS * 4) {
        int n = i >> 2, c = i & 3;
        x4[i] = pts[n * 5 + 1 + c];
    }
}

// ---------------------------------------------------------------------------
// KNN: per-batch brute force, LDS-tiled candidates, register top-16 insertion
// ---------------------------------------------------------------------------
__global__ __launch_bounds__(256)
void knn_kernel(const float* __restrict__ pts, int* __restrict__ idxout) {
    __shared__ float4 tile[256];
    const int n  = blockIdx.x * 256 + threadIdx.x;
    const float qx = pts[n * 5 + 1];
    const float qy = pts[n * 5 + 2];
    const float qz = pts[n * 5 + 3];
    const int   b  = (int)pts[n * 5 + 0];
    const int   start = b * (NPTS / BATCH);
    const float qd2 = qx * qx + qy * qy + qz * qz;

    float dist[KNBR];
    int   did[KNBR];
#pragma unroll
    for (int i = 0; i < KNBR; ++i) { dist[i] = __builtin_inff(); did[i] = 0; }

    for (int t = 0; t < NPTS / BATCH; t += 256) {
        const int c = start + t + threadIdx.x;
        float cx = pts[c * 5 + 1], cy = pts[c * 5 + 2], cz = pts[c * 5 + 3];
        tile[threadIdx.x] = make_float4(cx, cy, cz, cx * cx + cy * cy + cz * cz);
        __syncthreads();
        for (int j = 0; j < 256; ++j) {
            float4 p = tile[j];
            float d = qd2 + p.w - 2.0f * (qx * p.x + qy * p.y + qz * p.z);
            if (d < dist[KNBR - 1]) {
                dist[KNBR - 1] = d; did[KNBR - 1] = start + t + j;
#pragma unroll
                for (int s = KNBR - 1; s > 0; --s) {
                    if (dist[s] < dist[s - 1]) {
                        float td = dist[s]; dist[s] = dist[s - 1]; dist[s - 1] = td;
                        int   ti = did[s];  did[s]  = did[s - 1];  did[s - 1]  = ti;
                    }
                }
            }
        }
        __syncthreads();
    }
#pragma unroll
    for (int i = 0; i < KNBR; ++i) idxout[n * KNBR + i] = did[i];
}

// ---------------------------------------------------------------------------
// EdgeConv GEMM via V_WMMA_F32_16X16X32_F16.
// One wave per block; one 16x16 output tile per point n:
//   rows  = 16 neighbors (k), cols = 16 output channels, K-dim = 2*IN_C pad 32.
// PASS2=false: accumulate per-channel sum / sumsq partials (for BN stats).
// PASS2=true : apply BN scale/shift, ReLU, max over k -> xnext[n][out_c].
// ---------------------------------------------------------------------------
template <int IN_C, int OUT_C, bool PASS2>
__global__ __launch_bounds__(32)
void edgeconv_gemm(const float* __restrict__ x, const int* __restrict__ nbr,
                   const float* __restrict__ W, const float* __restrict__ bias,
                   const float* __restrict__ scale, const float* __restrict__ shift,
                   float* __restrict__ out) {
    constexpr int KD   = 2 * IN_C;           // 8, 32, 32, 64
    constexpr int KCH  = (KD + 31) / 32;     // 1 or 2 WMMA k-steps
    constexpr int KTOT = KCH * 32;
    constexpr int CT   = OUT_C / 16;         // 1 or 2 column tiles
    constexpr int IC4  = IN_C / 4;           // IN_C is always a multiple of 4

    __shared__ _Float16 els[16 * KTOT];

    const int lane = threadIdx.x;
    const int m  = lane & 15;   // row (k index) / column (channel) within tile
    const int hi = lane >> 4;

    const float4* __restrict__ xv = (const float4*)x;

    // --- B fragments (weights), loaded once; contiguous k order (A uses the same):
    //     lane: N = lane&15, halves s=0..15 -> K = (lane>>4)*16 + s (+ 32*kc)
    v16h Bf[CT][KCH];
#pragma unroll
    for (int ct = 0; ct < CT; ++ct) {
        const int col = ct * 16 + m;
#pragma unroll
        for (int kc = 0; kc < KCH; ++kc) {
#pragma unroll
            for (int s = 0; s < 16; ++s) {
                int k = kc * 32 + (hi << 4) + s;
                float w = (k < KD) ? W[col * KD + k] : 0.0f;
                Bf[ct][kc][s] = (_Float16)w;
            }
        }
    }

    float bs[CT], sc[CT], sh[CT], s1[CT], s2[CT];
#pragma unroll
    for (int ct = 0; ct < CT; ++ct) {
        const int ch = ct * 16 + m;
        bs[ct] = bias[ch];
        sc[ct] = PASS2 ? scale[ch] : 0.0f;
        sh[ct] = PASS2 ? shift[ch] : 0.0f;
        s1[ct] = 0.0f; s2[ct] = 0.0f;
    }

    for (int n = blockIdx.x; n < NPTS; n += GEMM_GRID) {
        // Build edge-feature tile e[16][KTOT] in LDS (f16), float4 gathers.
        // lanes hi=0: x_i half; lanes hi=1: (x_j - x_i) half + zero pad.
        const int j = nbr[n * KNBR + m];
        if (hi == 0) {
#pragma unroll
            for (int c4 = 0; c4 < IC4; ++c4) {
                float4 xi = xv[n * IC4 + c4];
                els[m * KTOT + c4 * 4 + 0] = (_Float16)xi.x;
                els[m * KTOT + c4 * 4 + 1] = (_Float16)xi.y;
                els[m * KTOT + c4 * 4 + 2] = (_Float16)xi.z;
                els[m * KTOT + c4 * 4 + 3] = (_Float16)xi.w;
            }
        } else {
#pragma unroll
            for (int c4 = 0; c4 < IC4; ++c4) {
                float4 xi = xv[n * IC4 + c4];
                float4 xj = xv[j * IC4 + c4];
                els[m * KTOT + IN_C + c4 * 4 + 0] = (_Float16)(xj.x - xi.x);
                els[m * KTOT + IN_C + c4 * 4 + 1] = (_Float16)(xj.y - xi.y);
                els[m * KTOT + IN_C + c4 * 4 + 2] = (_Float16)(xj.z - xi.z);
                els[m * KTOT + IN_C + c4 * 4 + 3] = (_Float16)(xj.w - xi.w);
            }
#pragma unroll
            for (int c = 2 * IN_C; c < KTOT; ++c)
                els[m * KTOT + c] = (_Float16)0.0f;
        }
        __syncthreads();

        // A fragments 16x32, same contiguous k order as B:
        //   row M = lane&15; half s -> K(in chunk) = 16*hi + s
        v16h Af[KCH];
#pragma unroll
        for (int kc = 0; kc < KCH; ++kc) {
#pragma unroll
            for (int s = 0; s < 16; ++s) {
                int k = kc * 32 + (hi << 4) + s;
                Af[kc][s] = els[m * KTOT + k];
            }
        }
        __syncthreads();  // safe to overwrite els next iteration

        v8f C[CT];
#pragma unroll
        for (int ct = 0; ct < CT; ++ct) {
            C[ct] = (v8f){0.f, 0.f, 0.f, 0.f, 0.f, 0.f, 0.f, 0.f};
#pragma unroll
            for (int kc = 0; kc < KCH; ++kc) {
                C[ct] = __builtin_amdgcn_wmma_f32_16x16x32_f16(
                    /*neg_a=*/false, Af[kc], /*neg_b=*/false, Bf[ct][kc],
                    /*c_mod=*/(short)0, C[ct], /*reuse_a=*/false, /*reuse_b=*/false);
            }
        }

        if (!PASS2) {
            // per-channel stats over this tile (lane's channel = ct*16 + m)
#pragma unroll
            for (int ct = 0; ct < CT; ++ct) {
                float ls = 0.0f, lq = 0.0f;
#pragma unroll
                for (int r = 0; r < 8; ++r) {
                    float v = C[ct][r] + bs[ct];
                    ls += v; lq += v * v;
                }
                s1[ct] += ls; s2[ct] += lq;
            }
        } else {
            // BN + ReLU + max over k (tile rows). Lane covers M = hi*8 + r;
            // combine halves with shfl_xor(16).
#pragma unroll
            for (int ct = 0; ct < CT; ++ct) {
                float vmax = 0.0f;  // ReLU output is >= 0
#pragma unroll
                for (int r = 0; r < 8; ++r) {
                    float v = (C[ct][r] + bs[ct]) * sc[ct] + sh[ct];
                    vmax = fmaxf(vmax, fmaxf(v, 0.0f));
                }
                float o = fmaxf(vmax, __shfl_xor(vmax, 16, 32));
                if (hi == 0) out[n * OUT_C + ct * 16 + m] = o;
            }
        }
    }

    if (!PASS2) {
        // deterministic partials: out[block][2*OUT_C] = {sum[ch], sumsq[ch]}
#pragma unroll
        for (int ct = 0; ct < CT; ++ct) {
            float t1 = s1[ct] + __shfl_xor(s1[ct], 16, 32);
            float t2 = s2[ct] + __shfl_xor(s2[ct], 16, 32);
            if (hi == 0) {
                out[blockIdx.x * (2 * OUT_C) + ct * 16 + m]         = t1;
                out[blockIdx.x * (2 * OUT_C) + OUT_C + ct * 16 + m] = t2;
            }
        }
    }
}

// ---------------------------------------------------------------------------
// Reduce partials -> BN scale/shift (single wave, fixed order: deterministic)
// ---------------------------------------------------------------------------
template <int OUT_C>
__global__ void reduce_stats(const float* __restrict__ partials,
                             const float* __restrict__ g, const float* __restrict__ t,
                             float* __restrict__ scale, float* __restrict__ shift) {
    int ch = threadIdx.x;
    if (ch >= OUT_C) return;
    float s = 0.0f, q = 0.0f;
    for (int b = 0; b < GEMM_GRID; ++b) {
        s += partials[b * (2 * OUT_C) + ch];
        q += partials[b * (2 * OUT_C) + OUT_C + ch];
    }
    const float cnt = (float)NPTS * (float)KNBR;
    float mu  = s / cnt;
    float var = q / cnt - mu * mu;
    float sc  = g[ch] * rsqrtf(var + EPS);
    scale[ch] = sc;
    shift[ch] = t[ch] - mu * sc;
}

// ---------------------------------------------------------------------------
extern "C" void kernel_launch(void* const* d_in, const int* in_sizes, int n_in,
                              void* d_out, int out_size, void* d_ws, size_t ws_size,
                              hipStream_t stream) {
    (void)in_sizes; (void)n_in; (void)out_size; (void)ws_size;
    const float* pts = (const float*)d_in[0];
    const float *Wp[4], *bp[4], *gp[4], *tp[4];
    for (int l = 0; l < 4; ++l) {
        Wp[l] = (const float*)d_in[1 + 4 * l];
        bp[l] = (const float*)d_in[2 + 4 * l];
        gp[l] = (const float*)d_in[3 + 4 * l];
        tp[l] = (const float*)d_in[4 + 4 * l];
    }

    char*  ws       = (char*)d_ws;
    int*   idxb     = (int*)ws;                         // N*16 ints
    float* xA       = (float*)(ws + (size_t)NPTS * KNBR * sizeof(int));
    float* xB       = xA + (size_t)NPTS * 32;
    float* partials = xB + (size_t)NPTS * 32;           // GRID * 64 floats
    float* scale    = partials + (size_t)GEMM_GRID * 64;
    float* shift    = scale + 32;
    float* outp     = (float*)d_out;

    extract_x<<<(NPTS * 4 + 255) / 256, 256, 0, stream>>>(pts, xA);
    knn_kernel<<<NPTS / 256, 256, 0, stream>>>(pts, idxb);

    // Layer 0: 4 -> 16   (xA -> xB)
    edgeconv_gemm<4, 16, false><<<GEMM_GRID, 32, 0, stream>>>(xA, idxb, Wp[0], bp[0], nullptr, nullptr, partials);
    reduce_stats<16><<<1, 16, 0, stream>>>(partials, gp[0], tp[0], scale, shift);
    edgeconv_gemm<4, 16, true ><<<GEMM_GRID, 32, 0, stream>>>(xA, idxb, Wp[0], bp[0], scale, shift, xB);

    // Layer 1: 16 -> 16  (xB -> xA)
    edgeconv_gemm<16, 16, false><<<GEMM_GRID, 32, 0, stream>>>(xB, idxb, Wp[1], bp[1], nullptr, nullptr, partials);
    reduce_stats<16><<<1, 16, 0, stream>>>(partials, gp[1], tp[1], scale, shift);
    edgeconv_gemm<16, 16, true ><<<GEMM_GRID, 32, 0, stream>>>(xB, idxb, Wp[1], bp[1], scale, shift, xA);

    // Layer 2: 16 -> 32  (xA -> xB)
    edgeconv_gemm<16, 32, false><<<GEMM_GRID, 32, 0, stream>>>(xA, idxb, Wp[2], bp[2], nullptr, nullptr, partials);
    reduce_stats<32><<<1, 32, 0, stream>>>(partials, gp[2], tp[2], scale, shift);
    edgeconv_gemm<16, 32, true ><<<GEMM_GRID, 32, 0, stream>>>(xA, idxb, Wp[2], bp[2], scale, shift, xB);

    // Layer 3: 32 -> 32  (xB -> d_out)
    edgeconv_gemm<32, 32, false><<<GEMM_GRID, 32, 0, stream>>>(xB, idxb, Wp[3], bp[3], nullptr, nullptr, partials);
    reduce_stats<32><<<1, 32, 0, stream>>>(partials, gp[3], tp[3], scale, shift);
    edgeconv_gemm<32, 32, true ><<<GEMM_GRID, 32, 0, stream>>>(xB, idxb, Wp[3], bp[3], scale, shift, outp);
}
